// NoBpttRnn_76510547411000
// MI455X (gfx1250) — matrix operand order repaired and checked by reading, not compiled, not run on hardware
//
#include <hip/hip_runtime.h>
#include <math.h>

// ---------------------------------------------------------------------------
// CDNA5 WMMA bf16: D(f32 16x16) = A(bf16 16x32) * B(bf16 32x16) + C
// ---------------------------------------------------------------------------
typedef __attribute__((ext_vector_type(16))) __bf16        v16bf;
typedef __attribute__((ext_vector_type(8)))  float         v8f;
typedef __attribute__((ext_vector_type(4)))  unsigned int  v4u;
typedef __attribute__((ext_vector_type(8)))  int           v8i;
typedef __attribute__((ext_vector_type(4)))  int           v4i;

union Frag512 { v16bf v; uint4 q[2]; };   // 32 bytes per lane
union CFrag   { v8f   v; float f[8]; };

__device__ __forceinline__ __bf16 to_bf16(float f) { return (__bf16)f; }

// A fragment (16x32 bf16) from row-major f32 (inline cvt -> v_cvt_pk_bf16_f32).
__device__ __forceinline__ v16bf load_a_frag_f32(const float* __restrict__ src,
                                                 int ld, int mBase, int kBase,
                                                 int lane) {
    const int row = mBase + (lane & 15);
    const int kc0 = kBase + ((lane < 16) ? 0 : 8);
    const float* p = src + (size_t)row * ld + kc0;
    const float4 f0 = *(const float4*)(p + 0);
    const float4 f1 = *(const float4*)(p + 4);
    const float4 f2 = *(const float4*)(p + 16);
    const float4 f3 = *(const float4*)(p + 20);
    v16bf a;
    a[0]  = to_bf16(f0.x); a[1]  = to_bf16(f0.y);
    a[2]  = to_bf16(f0.z); a[3]  = to_bf16(f0.w);
    a[4]  = to_bf16(f1.x); a[5]  = to_bf16(f1.y);
    a[6]  = to_bf16(f1.z); a[7]  = to_bf16(f1.w);
    a[8]  = to_bf16(f2.x); a[9]  = to_bf16(f2.y);
    a[10] = to_bf16(f2.z); a[11] = to_bf16(f2.w);
    a[12] = to_bf16(f3.x); a[13] = to_bf16(f3.y);
    a[14] = to_bf16(f3.z); a[15] = to_bf16(f3.w);
    return a;
}

// Pre-swizzled 512-half fragment (A or B block): 32 contiguous bytes per lane.
// Works for global or LDS-derived pointers (ds_load_b128 when src is shared).
__device__ __forceinline__ v16bf load_frag512(const __bf16* buf,
                                              int blockIdxLinear, int lane) {
    const __bf16* p = buf + (size_t)blockIdxLinear * 512 + lane * 16;
    Frag512 f;
    f.q[0] = *(const uint4*)(p);
    f.q[1] = *(const uint4*)(p + 8);
    return f.v;
}

// ---------------------------------------------------------------------------
// TDM: 1-D async copy of nbytes (multiple of 8) global -> LDS.
// D# group0: count=1 | lds_addr | global_addr(57b) | type=2.
// D# group1: data_size=3 (8B), tensor_dim0 = tile_dim0 = n8, stride0 = n8.
// Tracked by TENSORcnt.
// ---------------------------------------------------------------------------
__device__ __forceinline__ void tdm_load_1d(unsigned lds_byte_addr,
                                            const void* gsrc,
                                            unsigned n8 /* 8-byte elems */) {
    const unsigned long long ga = (unsigned long long)gsrc;
    v4u g0;
    g0.x = 1u;                                            // count=1 (valid)
    g0.y = lds_byte_addr;                                 // LDS dest (bytes)
    g0.z = (unsigned)(ga & 0xFFFFFFFFu);                  // global_addr[31:0]
    g0.w = (unsigned)((ga >> 32) & 0x1FFFFFFu) | (2u << 30); // [56:32] | type=2
    v8i g1;
    g1[0] = 0x30000;                                      // data_size=3 (8B)
    g1[1] = (int)((n8 & 0xFFFFu) << 16);                  // tensor_dim0 lo16
    g1[2] = (int)(((n8 >> 16) & 0xFFFFu) | (1u << 16));   // td0 hi | tensor_dim1=1
    g1[3] = (int)((n8 & 0xFFFFu) << 16);                  // tile_dim0
    g1[4] = 0;                                            // tile_dim1/2 unused
    g1[5] = (int)n8;                                      // tensor_dim0_stride
    g1[6] = 0;
    g1[7] = 0;
    v4i g2 = {0, 0, 0, 0};
    v4i g3 = {0, 0, 0, 0};
#if __clang_major__ >= 23
    v8i g4 = {0, 0, 0, 0, 0, 0, 0, 0};
    __builtin_amdgcn_tensor_load_to_lds(g0, g1, g2, g3, g4, 0);
#else
    __builtin_amdgcn_tensor_load_to_lds(g0, g1, g2, g3, 0);
#endif
}

// ---------------------------------------------------------------------------
// Convert row-major f32 KxN weights into bf16 WMMA-B blocks.
// ---------------------------------------------------------------------------
__global__ void convert_w_kernel(const float* __restrict__ W,
                                 __bf16* __restrict__ out,
                                 int K, int N) {
    int tid = blockIdx.x * blockDim.x + threadIdx.x;
    if (tid >= K * N) return;
    int blk  = tid >> 9;
    int r    = tid & 511;
    int lane = r >> 4;
    int j    = r & 15;
    int ntiles = N >> 4;
    int kt = blk / ntiles;
    int nt = blk % ntiles;
    int k = kt * 32 + ((lane < 16) ? 0 : 8) + (j & 7) + ((j & 8) << 1);
    int n = nt * 16 + (lane & 15);
    out[tid] = to_bf16(W[(size_t)k * N + n]);
}

// ---------------------------------------------------------------------------
// Kernel 1: out[m,:] = x[m,:] @ Wih + b. One wave = 32x64 strip, 8 WMMA/k-tile.
// ---------------------------------------------------------------------------
__global__ __launch_bounds__(256)
void xw_gemm_kernel(const float* __restrict__ x,
                    const __bf16* __restrict__ wihf,
                    const float* __restrict__ bias,
                    float* __restrict__ out,
                    int M, int K, int N) {
    const int lane = threadIdx.x & 31;
    const int wid  = (blockIdx.x * blockDim.x + threadIdx.x) >> 5;
    const int ntiles    = N >> 4;       // 64
    const int colGroups = ntiles >> 2;  // 16
    const int mBase = (wid / colGroups) * 32;
    const int nt0   = (wid % colGroups) * 4;

    v8f acc[2][4];
#pragma unroll
    for (int r = 0; r < 2; ++r)
#pragma unroll
        for (int c = 0; c < 4; ++c) acc[r][c] = (v8f){};

    const int kTiles = K >> 5;
    for (int kt = 0; kt < kTiles; ++kt) {
        if (kt + 1 < kTiles) {
            __builtin_prefetch(wihf + (size_t)((kt + 1) * ntiles + nt0) * 512, 0, 1);
        }
        v16bf a0 = load_a_frag_f32(x, K, mBase,      kt * 32, lane);
        v16bf a1 = load_a_frag_f32(x, K, mBase + 16, kt * 32, lane);
        v16bf b[4];
#pragma unroll
        for (int c = 0; c < 4; ++c)
            b[c] = load_frag512(wihf, kt * ntiles + nt0 + c, lane);
#pragma unroll
        for (int c = 0; c < 4; ++c) {
            acc[0][c] = __builtin_amdgcn_wmma_f32_16x16x32_bf16(
                false, a0, false, b[c], (short)0, acc[0][c], false, false);
            acc[1][c] = __builtin_amdgcn_wmma_f32_16x16x32_bf16(
                false, a1, false, b[c], (short)0, acc[1][c], false, false);
        }
    }

    const int rOff = (lane < 16) ? 0 : 8;
#pragma unroll
    for (int r = 0; r < 2; ++r) {
#pragma unroll
        for (int c = 0; c < 4; ++c) {
            const int col = (nt0 + c) * 16 + (lane & 15);
            const float bv = bias[col];
            CFrag cf; cf.v = acc[r][c];
#pragma unroll
            for (int i = 0; i < 8; ++i) {
                out[(size_t)(mBase + r * 16 + rOff + i) * N + col] = cf.f[i] + bv;
            }
        }
    }
}

// ---------------------------------------------------------------------------
// Kernel 2 (launched S times): h_t = tanh(xw_t + h_{t-1} @ Whh).
// Block = 64 rows x 128 cols. Whh k-slabs (8KB, contiguous in fragment layout)
// are DMA'd into LDS by the Tensor Data Mover, double-buffered, and consumed
// via ds_load_b128. h is kept bf16 in A-fragment layout (no cvt in mainloop).
// Wave w: row-tile mt = w>>1, col tiles (w&1)*4 .. +3 within the block's 8.
// ---------------------------------------------------------------------------
__global__ __launch_bounds__(256)
void rnn_step_kernel(const __bf16* __restrict__ hprev,  // [4][32][512] A-frags
                     __bf16* __restrict__ hnext,
                     const __bf16* __restrict__ whhf,
                     float* __restrict__ out,
                     int t, int S, int H) {
    const int lane    = threadIdx.x & 31;
    const int w       = threadIdx.x >> 5;   // 0..7
    const int mt      = w >> 1;             // 0..3
    const int nt0loc  = (w & 1) * 4;        // 0 or 4 (within block's 8 tiles)
    const int ntBlock = blockIdx.x * 8;     // first col tile of this block
    const int ntiles  = H >> 4;             // 64
    const int kTiles  = H >> 5;             // 32

    // Double-buffered B slab: 8 tiles * 512 halves * 2B = 8KB per buffer.
    __shared__ __align__(16) __bf16 bsl[2][8 * 512];

    v8f acc[4];
#pragma unroll
    for (int c = 0; c < 4; ++c) acc[c] = (v8f){};

    // Prologue: TDM for k-tile 0 into buffer 0.
    if (w == 0) {
        tdm_load_1d((unsigned)(uintptr_t)&bsl[0][0],
                    whhf + (size_t)(0 * ntiles + ntBlock) * 512, 1024);
    }

    for (int kt = 0; kt < kTiles; ++kt) {
        const int slot = kt & 1;
        if (w == 0) {
            if (kt + 1 < kTiles) {
                tdm_load_1d((unsigned)(uintptr_t)&bsl[slot ^ 1][0],
                            whhf + (size_t)((kt + 1) * ntiles + ntBlock) * 512,
                            1024);
                __builtin_amdgcn_s_wait_tensorcnt(1);  // slab kt complete
            } else {
                __builtin_amdgcn_s_wait_tensorcnt(0);
            }
        }
        __syncthreads();  // slab `slot` visible to all waves

        v16bf a = load_frag512(hprev, mt * kTiles + kt, lane);
#pragma unroll
        for (int c = 0; c < 4; ++c) {
            v16bf b = load_frag512(&bsl[slot][0], nt0loc + c, lane);
            acc[c] = __builtin_amdgcn_wmma_f32_16x16x32_bf16(
                false, a, false, b, (short)0, acc[c], false, false);
        }
        __syncthreads();  // reads done before TDM overwrites slot^1 next iter
    }

    // Epilogue: add xw (in-place in out), tanh, store f32 to out and bf16
    // (A-fragment-swizzled) to hnext for the next timestep.
    const int rOff = (lane < 16) ? 0 : 8;
#pragma unroll
    for (int c = 0; c < 4; ++c) {
        const int col = (ntBlock + nt0loc + c) * 16 + (lane & 15);
        const int kt2   = col >> 5;
        const int cc    = col & 31;
        const int laneH = (cc & 8) << 1;               // 0 or 16
        const int j     = (cc & 7) | ((cc & 16) >> 1); // element 0..15
        __bf16* hdst = hnext + (size_t)(mt * kTiles + kt2) * 512 + j;
        CFrag cf; cf.v = acc[c];
#pragma unroll
        for (int i = 0; i < 8; ++i) {
            const int r    = rOff + i;
            const int brow = mt * 16 + r;
            const size_t o = ((size_t)brow * S + t) * H + col;
            const float h = tanhf(out[o] + cf.f[i]);
            out[o] = h;
            hdst[(r | laneH) * 16] = to_bf16(h);
        }
    }
}

// ---------------------------------------------------------------------------
// Host launcher
// ---------------------------------------------------------------------------
extern "C" void kernel_launch(void* const* d_in, const int* in_sizes, int n_in,
                              void* d_out, int out_size, void* d_ws, size_t ws_size,
                              hipStream_t stream) {
    const float* x    = (const float*)d_in[0];   // [B,S,D]
    const float* Wih  = (const float*)d_in[1];   // [D,H]
    const float* Whh  = (const float*)d_in[2];   // [H,H]
    const float* bias = (const float*)d_in[3];   // [H]
    float* out = (float*)d_out;                  // [B,S,H]

    const int B = 64, S = 512, D = 1024, H = 1024;
    const int M = B * S;

    // Workspace: [Wih frags 2MB | Whh frags 2MB | h ping-pong bf16 256KB]
    __bf16* wihf = (__bf16*)d_ws;
    __bf16* whhf = wihf + (size_t)D * H;
    __bf16* hbuf = whhf + (size_t)H * H;

    // 1) One-time weight conversion into WMMA-B-fragment layout.
    {
        const int total = D * H;
        convert_w_kernel<<<(total + 255) / 256, 256, 0, stream>>>(Wih, wihf, D, H);
        convert_w_kernel<<<(total + 255) / 256, 256, 0, stream>>>(Whh, whhf, H, H);
    }

    // 2) Input projection GEMM straight into d_out (acts as xw buffer).
    {
        const int waves = (M / 32) * (H / 64);   // 16384 waves
        xw_gemm_kernel<<<waves / 8, 256, 0, stream>>>(x, wihf, bias, out, M, D, H);
    }

    // 3) h0 = 0 (bf16 zeros == 0x0000), then 512 dependent steps.
    hipMemsetAsync(hbuf, 0, (size_t)B * H * sizeof(__bf16), stream);
    for (int t = 0; t < S; ++t) {
        __bf16* hprev = hbuf + (size_t)(t & 1) * B * H;
        __bf16* hnext = hbuf + (size_t)((t + 1) & 1) * B * H;
        // 8 blocks x 256 threads: each block = 64 rows x 128 cols
        rnn_step_kernel<<<8, 256, 0, stream>>>(hprev, hnext, whhf, out, t, S, H);
    }
}
